// MR_GNN_35854386987430
// MI455X (gfx1250) — matrix operand order repaired and checked by reading, not compiled, not run on hardware
//
#include <hip/hip_runtime.h>

#define DIM  128
#define ODIM 64
#define NREL 8

typedef __attribute__((ext_vector_type(16))) __bf16 v16bf;
typedef __attribute__((ext_vector_type(8)))  __bf16 v8bf;
typedef __attribute__((ext_vector_type(8)))  float  v8f;

// ---------------- elementwise helpers ----------------

__global__ void fill_f32(float* __restrict__ p, float v, int n) {
  int i = blockIdx.x * blockDim.x + threadIdx.x;
  if (i < n) p[i] = v;
}

__global__ void deg_count(const int* __restrict__ dst, float* __restrict__ deg, int E) {
  int e = blockIdx.x * blockDim.x + threadIdx.x;
  if (e < E) atomicAdd(&deg[dst[e]], 1.0f);
}

// SCALED=0: dst[i] = bf16(src[i]);  SCALED=1: dst[i] = bf16(src[i]/max(deg[i/128],1))
template <int SCALED>
__global__ void to_bf16(const float* __restrict__ src, const float* __restrict__ deg,
                        __bf16* __restrict__ dst, int n) {
  int i = blockIdx.x * blockDim.x + threadIdx.x;
  if (i >= n) return;
  float v = src[i];
  if (SCALED) v /= fmaxf(deg[i >> 7], 1.0f);
  dst[i] = (__bf16)v;
}

// h[i] = relu(h[i] / max(deg,1))   (in place)
__global__ void relu_scale_deg(float* __restrict__ h, const float* __restrict__ deg, int n) {
  int i = blockIdx.x * blockDim.x + threadIdx.x;
  if (i >= n) return;
  float d = fmaxf(deg[i >> 7], 1.0f);
  h[i] = fmaxf(h[i] / d, 0.0f);
}

// Wt[n*K + k] = bf16(W[k*Nn + n]) : store B transposed so a WMMA B column is contiguous
__global__ void weight_t_bf16(const float* __restrict__ W, __bf16* __restrict__ Wt,
                              int K, int Nn) {
  int i = blockIdx.x * blockDim.x + threadIdx.x;
  if (i >= K * Nn) return;
  int n = i / K, k = i - n * K;
  Wt[i] = (__bf16)W[(size_t)k * Nn + n];
}

// ---------------- bf16 WMMA GEMM: C[M,NN] = A[M,128] * Bt[NN,128]^T (+epilogue) -------------
// MODE 0: C = acc
// MODE 1: C = Hprev + alpha*(acc + bias)      (group stage)
// MODE 2: C = acc + bias                      (output layer)
// M must be a multiple of 16 (50000 = 3125*16); NN is 128 or 64 (compile-time).
template <int MODE, int NN>
__global__ __launch_bounds__(128)
void gemm_bf16_wmma(const __bf16* __restrict__ A, const __bf16* __restrict__ Bt,
                    float* __restrict__ C, int M,
                    const float* __restrict__ bias, const float* __restrict__ Hprev,
                    const float* __restrict__ alpha_p) {
  const int K = DIM;
  int wave = threadIdx.x >> 5;
  int lane = threadIdx.x & 31;
  int m0 = (blockIdx.x * 4 + wave) * 16;
  if (m0 >= M) return;                 // wave-uniform exit; EXEC all-ones inside
  int lh  = lane >> 4;                 // lane half (K sub-block select)
  int l15 = lane & 15;

  // A fragment: lane l (0..15) holds row m0+l; elems 0..15 = K0+16h..K0+16h+15
  const __bf16* ap = A + (size_t)(m0 + l15) * K + 16 * lh;
  v16bf a[4];
#pragma unroll
  for (int kt = 0; kt < 4; ++kt) {
    v8bf lo = *(const v8bf*)(ap + kt * 32);
    v8bf hi = *(const v8bf*)(ap + kt * 32 + 8);
    v16bf af;
#pragma unroll
    for (int j = 0; j < 8; ++j) { af[j] = lo[j]; af[8 + j] = hi[j]; }
    a[kt] = af;
  }

  float alpha = 0.0f;
  if (MODE == 1) alpha = alpha_p[0];

#pragma unroll
  for (int n0 = 0; n0 < NN; n0 += 16) {
    int col = n0 + l15;
    // B fragment: lane l holds column n0+l, 16 consecutive K values (Bt row-major in K)
    const __bf16* bp = Bt + (size_t)col * K + 16 * lh;
    v8f acc = {};
#pragma unroll
    for (int kt = 0; kt < 4; ++kt) {
      v16bf b = *(const v16bf*)(bp + kt * 32);
      acc = __builtin_amdgcn_wmma_f32_16x16x32_bf16(false, a[kt], false, b,
                                                    (short)0, acc, false, false);
    }
    float bv = 0.0f;
    if (MODE != 0) bv = bias[col];
#pragma unroll
    for (int j = 0; j < 8; ++j) {
      int row = m0 + j + 8 * lh;       // C layout: VGPR j, lanes0-15 -> M=j, lanes16-31 -> M=8+j
      size_t idx = (size_t)row * NN + col;
      float v = acc[j];
      if (MODE == 1)      v = Hprev[idx] + alpha * (v + bv);
      else if (MODE == 2) v = v + bv;
      C[idx] = v;
    }
  }
}

// ---------------- edge gather + scatter-add: Acc[dst] += X[src] (optional relation filter) ----
template <int FILTERED>
__global__ __launch_bounds__(128)
void edge_scatter_add(const int* __restrict__ src, const int* __restrict__ dst,
                      const int* __restrict__ etype, int rel,
                      const float* __restrict__ X, float* __restrict__ Acc, int E) {
  int e = blockIdx.x * 4 + (threadIdx.x >> 5);   // one wave per edge
  if (e >= E) return;
  if (FILTERED && etype[e] != rel) return;
  int lane = threadIdx.x & 31;
  int s = src[e], d = dst[e];
  float4 v = ((const float4*)(X + (size_t)s * DIM))[lane];   // 512B coalesced row
  float* out = Acc + (size_t)d * DIM + lane * 4;
  atomicAdd(out + 0, v.x);
  atomicAdd(out + 1, v.y);
  atomicAdd(out + 2, v.z);
  atomicAdd(out + 3, v.w);
}

// ---------------- host orchestration ----------------

extern "C" void kernel_launch(void* const* d_in, const int* in_sizes, int n_in,
                              void* d_out, int out_size, void* d_ws, size_t ws_size,
                              hipStream_t stream) {
  (void)n_in; (void)out_size; (void)ws_size;
  const float* x    = (const float*)d_in[0];
  const int*   ei   = (const int*)d_in[1];
  const int*   etyp = (const int*)d_in[2];
  const float* W1   = (const float*)d_in[3];
  const float* W01  = (const float*)d_in[4];
  const float* a1   = (const float*)d_in[5];
  const float* P1   = (const float*)d_in[6];
  const float* b1   = (const float*)d_in[7];
  const float* W2   = (const float*)d_in[8];
  const float* W02  = (const float*)d_in[9];
  const float* a2   = (const float*)d_in[10];
  const float* P2   = (const float*)d_in[11];
  const float* b2   = (const float*)d_in[12];
  const float* OW   = (const float*)d_in[13];
  const float* Ob   = (const float*)d_in[14];

  const int N = in_sizes[0] / DIM;     // 50000
  const int E = in_sizes[1] / 2;       // 1600000
  const int* src = ei;
  const int* dst = ei + E;

  char* w = (char*)d_ws;
  auto carve = [&](size_t bytes) -> void* {
    void* p = (void*)w;
    w += (bytes + 255) & ~(size_t)255;
    return p;
  };
  float*  bufA = (float*)carve((size_t)N * DIM * 4);   // h_acc / post-relu h
  float*  bufB = (float*)carve((size_t)N * DIM * 4);   // per-relation xW_r
  float*  bufC = (float*)carve((size_t)N * DIM * 4);   // group aggregation
  float*  bufD = (float*)carve((size_t)N * DIM * 4);   // group output
  float*  deg  = (float*)carve((size_t)N * 4);
  __bf16* xb   = (__bf16*)carve((size_t)N * DIM * 2);  // bf16 activations
  const size_t WG = (size_t)DIM * DIM;
  __bf16* W1t  = (__bf16*)carve(NREL * WG * 2);
  __bf16* W2t  = (__bf16*)carve(NREL * WG * 2);
  __bf16* W01t = (__bf16*)carve(WG * 2);
  __bf16* W02t = (__bf16*)carve(WG * 2);
  __bf16* P1t  = (__bf16*)carve(WG * 2);
  __bf16* P2t  = (__bf16*)carve(WG * 2);
  __bf16* OWt  = (__bf16*)carve((size_t)DIM * ODIM * 2);

  const int ND       = N * DIM;
  const int gridND   = (ND + 255) / 256;
  const int mtiles   = (N + 15) / 16;
  const int gemmGrid = (mtiles + 3) / 4;
  const int edgeGrid = (E + 3) / 4;
  const int gridW    = ((int)WG + 255) / 256;

  // --- weights -> bf16, transposed ---
  for (int r = 0; r < NREL; ++r) {
    weight_t_bf16<<<gridW, 256, 0, stream>>>(W1 + r * WG, W1t + r * WG, DIM, DIM);
    weight_t_bf16<<<gridW, 256, 0, stream>>>(W2 + r * WG, W2t + r * WG, DIM, DIM);
  }
  weight_t_bf16<<<gridW, 256, 0, stream>>>(W01, W01t, DIM, DIM);
  weight_t_bf16<<<gridW, 256, 0, stream>>>(W02, W02t, DIM, DIM);
  weight_t_bf16<<<gridW, 256, 0, stream>>>(P1,  P1t,  DIM, DIM);
  weight_t_bf16<<<gridW, 256, 0, stream>>>(P2,  P2t,  DIM, DIM);
  weight_t_bf16<<<(DIM * ODIM + 255) / 256, 256, 0, stream>>>(OW, OWt, DIM, ODIM);

  // --- degree (shared by both layers and both group stages) ---
  fill_f32<<<(N + 255) / 256, 256, 0, stream>>>(deg, 0.0f, N);
  deg_count<<<(E + 255) / 256, 256, 0, stream>>>(dst, deg, E);

  // --- layer 1 RGCN ---
  to_bf16<0><<<gridND, 256, 0, stream>>>(x, nullptr, xb, ND);
  gemm_bf16_wmma<0, DIM><<<gemmGrid, 128, 0, stream>>>(xb, W01t, bufA, N, nullptr, nullptr, nullptr);
  for (int r = 0; r < NREL; ++r) {
    gemm_bf16_wmma<0, DIM><<<gemmGrid, 128, 0, stream>>>(xb, W1t + r * WG, bufB, N,
                                                         nullptr, nullptr, nullptr);
    edge_scatter_add<1><<<edgeGrid, 128, 0, stream>>>(src, dst, etyp, r, bufB, bufA, E);
  }
  relu_scale_deg<<<gridND, 256, 0, stream>>>(bufA, deg, ND);

  // --- group 1 ---
  fill_f32<<<gridND, 256, 0, stream>>>(bufC, 0.0f, ND);
  edge_scatter_add<0><<<edgeGrid, 128, 0, stream>>>(src, dst, etyp, -1, bufA, bufC, E);
  to_bf16<1><<<gridND, 256, 0, stream>>>(bufC, deg, xb, ND);
  gemm_bf16_wmma<1, DIM><<<gemmGrid, 128, 0, stream>>>(xb, P1t, bufD, N, b1, bufA, a1);

  // --- layer 2 RGCN ---
  to_bf16<0><<<gridND, 256, 0, stream>>>(bufD, nullptr, xb, ND);
  gemm_bf16_wmma<0, DIM><<<gemmGrid, 128, 0, stream>>>(xb, W02t, bufA, N, nullptr, nullptr, nullptr);
  for (int r = 0; r < NREL; ++r) {
    gemm_bf16_wmma<0, DIM><<<gemmGrid, 128, 0, stream>>>(xb, W2t + r * WG, bufB, N,
                                                         nullptr, nullptr, nullptr);
    edge_scatter_add<1><<<edgeGrid, 128, 0, stream>>>(src, dst, etyp, r, bufB, bufA, E);
  }
  relu_scale_deg<<<gridND, 256, 0, stream>>>(bufA, deg, ND);

  // --- group 2 ---
  fill_f32<<<gridND, 256, 0, stream>>>(bufC, 0.0f, ND);
  edge_scatter_add<0><<<edgeGrid, 128, 0, stream>>>(src, dst, etyp, -1, bufA, bufC, E);
  to_bf16<1><<<gridND, 256, 0, stream>>>(bufC, deg, xb, ND);
  gemm_bf16_wmma<1, DIM><<<gemmGrid, 128, 0, stream>>>(xb, P2t, bufD, N, b2, bufA, a2);

  // --- output projection ---
  to_bf16<0><<<gridND, 256, 0, stream>>>(bufD, nullptr, xb, ND);
  gemm_bf16_wmma<2, ODIM><<<gemmGrid, 128, 0, stream>>>(xb, OWt, (float*)d_out, N,
                                                        Ob, nullptr, nullptr);
}
